// EncoderBlock_67104569032840
// MI455X (gfx1250) — matrix operand order repaired and checked by reading, not compiled, not run on hardware
//
#include <hip/hip_runtime.h>
#include <math.h>

#define B_   32
#define L_   512
#define D_   512
#define H_   8
#define DH_  64
#define KW   7
#define LD_  (L_ * D_)          // 262144
#define BLD  ((size_t)B_ * L_ * D_)
#define NEGV (-1e30f)

typedef __attribute__((ext_vector_type(16))) _Float16 v16h;
typedef __attribute__((ext_vector_type(8)))  _Float16 v8h;
typedef __attribute__((ext_vector_type(4)))  _Float16 v4h;
typedef __attribute__((ext_vector_type(8)))  float    v8f;

// ---------------------------------------------------------------- fragments
__device__ __forceinline__ int laneid() { return threadIdx.x & 31; }

// A-fragment (16x32 f16, M x K). Per ISA 7.12.2: lane<16 -> M=lane, K in
// {kb..kb+7, kb+16..kb+23} with kb = (lane>=16)*8.
__device__ __forceinline__ v16h frag_a_ld(const _Float16* base, int ld) {
  int l = laneid();
  const _Float16* p = base + (l & 15) * ld + ((l >> 4) << 3);
  v8h lo = *(const v8h*)(p);
  v8h hi = *(const v8h*)(p + 16);
  return __builtin_shufflevector(lo, hi, 0,1,2,3,4,5,6,7,8,9,10,11,12,13,14,15);
}

// B-fragment (32x16 f16, K x N) read from an N-major LDS tile base[n*ld + k].
// lane<16 -> N=lane, K = 0..7,16..23; lane>=16 -> N=lane-16, K = 8..15,24..31.
__device__ __forceinline__ v16h frag_b_ld(const _Float16* base, int ld) {
  int l = laneid();
  const _Float16* p = base + (l & 15) * ld + ((l >> 4) << 3);
  v8h lo = *(const v8h*)(p);
  v8h hi = *(const v8h*)(p + 16);
  return __builtin_shufflevector(lo, hi, 0,1,2,3,4,5,6,7,8,9,10,11,12,13,14,15);
}

__device__ __forceinline__ v8f wmma32(v16h a, v16h b, v8f c) {
  return __builtin_amdgcn_wmma_f32_16x16x32_f16(false, a, false, b, (short)0, c,
                                                false, false);
}

__device__ __forceinline__ v4h cvt4(float4 v) {
  v4h h;
  h[0] = (_Float16)v.x; h[1] = (_Float16)v.y;
  h[2] = (_Float16)v.z; h[3] = (_Float16)v.w;
  return h;
}

// ---------------------------------------------------------------- GEMM
// C[M,N] = act(A[M,K] fp32 * Bp[N,K] f16 + bias), 128x128 tile, BK=32,
// double-buffered LDS, vectorized 16B staging.
#define LDA_S 40
#define LDB_S 40

struct RegsAB { float4 ra[4]; v8h rb[2]; };

__device__ __forceinline__ void gemm_gload(RegsAB& rg, const float* __restrict__ A,
                                           const _Float16* __restrict__ Bp,
                                           int bm, int bn, int Kd, int k0, int t) {
#pragma unroll
  for (int i = 0; i < 4; ++i) {           // A: 128x32 fp32, float4 per lane
    int vi = t + i * 256;
    int r = vi >> 3, c = (vi & 7) << 2;
    rg.ra[i] = *(const float4*)&A[(size_t)(bm + r) * Kd + k0 + c];
  }
#pragma unroll
  for (int i = 0; i < 2; ++i) {           // B: 128(n) x 32(k) f16, v8h per lane
    int vi = t + i * 256;
    int n = vi >> 2, k = (vi & 3) << 3;
    rg.rb[i] = *(const v8h*)&Bp[(size_t)(bn + n) * Kd + k0 + k];
  }
}

__device__ __forceinline__ void gemm_sstore(const RegsAB& rg,
                                            _Float16* As, _Float16* Bs, int t) {
#pragma unroll
  for (int i = 0; i < 4; ++i) {
    int vi = t + i * 256;
    int r = vi >> 3, c = (vi & 7) << 2;
    *(v4h*)&As[r * LDA_S + c] = cvt4(rg.ra[i]);
  }
#pragma unroll
  for (int i = 0; i < 2; ++i) {
    int vi = t + i * 256;
    int n = vi >> 2, k = (vi & 3) << 3;
    *(v8h*)&Bs[n * LDB_S + k] = rg.rb[i];
  }
}

__global__ __launch_bounds__(256) void k_gemm(
    const float* __restrict__ A, const _Float16* __restrict__ Bp,
    const float* __restrict__ bias, float* __restrict__ C,
    int M, int N, int Kd, int relu) {
  __shared__ __align__(16) _Float16 As[2][128 * LDA_S];
  __shared__ __align__(16) _Float16 Bs[2][128 * LDB_S];
  int t = threadIdx.x;
  int wave = t >> 5;
  int wm = wave & 1, wn = wave >> 1;           // 2 x 4 wave grid
  int bm = blockIdx.x * 128, bn = blockIdx.y * 128;

  v8f zero = {};
  v8f acc[4][2];
#pragma unroll
  for (int mt = 0; mt < 4; ++mt)
#pragma unroll
    for (int nt = 0; nt < 2; ++nt) acc[mt][nt] = zero;

  RegsAB rg;
  gemm_gload(rg, A, Bp, bm, bn, Kd, 0, t);
  gemm_sstore(rg, As[0], Bs[0], t);
  __syncthreads();

  int nIter = Kd >> 5;
  for (int it = 0; it < nIter; ++it) {
    int buf = it & 1;
    bool haveNext = (it + 1 < nIter);
    if (haveNext) gemm_gload(rg, A, Bp, bm, bn, Kd, (it + 1) << 5, t);

    v16h af[4], bf[2];
#pragma unroll
    for (int mt = 0; mt < 4; ++mt)
      af[mt] = frag_a_ld(&As[buf][(wm * 64 + mt * 16) * LDA_S], LDA_S);
#pragma unroll
    for (int nt = 0; nt < 2; ++nt)
      bf[nt] = frag_b_ld(&Bs[buf][(wn * 32 + nt * 16) * LDB_S], LDB_S);
#pragma unroll
    for (int mt = 0; mt < 4; ++mt)
#pragma unroll
      for (int nt = 0; nt < 2; ++nt)
        acc[mt][nt] = wmma32(af[mt], bf[nt], acc[mt][nt]);

    if (haveNext) {
      gemm_sstore(rg, As[buf ^ 1], Bs[buf ^ 1], t);
      __syncthreads();
    }
  }

  int l = laneid();
  int cn = l & 15, rbase = (l >> 4) * 8;
#pragma unroll
  for (int nt = 0; nt < 2; ++nt) {
    int n = bn + wn * 32 + nt * 16 + cn;
    float bv = bias ? bias[n] : 0.f;
#pragma unroll
    for (int mt = 0; mt < 4; ++mt)
#pragma unroll
      for (int r = 0; r < 8; ++r) {
        int m = bm + wm * 64 + mt * 16 + rbase + r;
        float v = acc[mt][nt][r] + bv;
        if (relu) v = fmaxf(v, 0.f);
        C[(size_t)m * N + n] = v;
      }
  }
}

// ---------------------------------------------------------------- flash attn
// grid: x = B*H, y = L/128. Block = 8 waves, each owns 16 query rows.
__global__ __launch_bounds__(256) void k_flash(
    const float* __restrict__ Q, const float* __restrict__ Kx,
    const float* __restrict__ Vx, const float* __restrict__ mask,
    float* __restrict__ O) {
  __shared__ __align__(16) _Float16 Ks[64 * 72];       // [j][dh]
  __shared__ __align__(16) _Float16 VsT[64 * 72];      // [dh][j]
  __shared__ __align__(16) _Float16 Ws[8][16 * 72];    // per-wave Q then P

  int bh = blockIdx.x;
  int b = bh >> 3, h = bh & 7;
  int t = threadIdx.x, wave = t >> 5, lane = t & 31;
  int m0 = blockIdx.y * 128 + wave * 16;
  const float scale = 0.125f;                          // 1/sqrt(64)
  size_t baseQ = (size_t)b * L_ * D_ + (size_t)h * DH_;

  _Float16* Qs = Ws[wave];
#pragma unroll
  for (int i = 0; i < 8; ++i) {                        // 16x64 Q, pre-scaled
    int vi = lane + i * 32;
    int r = vi >> 4, c = (vi & 15) << 2;
    float4 qv = *(const float4*)&Q[baseQ + (size_t)(m0 + r) * D_ + c];
    qv.x *= scale; qv.y *= scale; qv.z *= scale; qv.w *= scale;
    *(v4h*)&Qs[r * 72 + c] = cvt4(qv);
  }
  asm volatile("s_wait_dscnt 0" ::: "memory");
  __builtin_amdgcn_wave_barrier();
  v16h qa0 = frag_a_ld(Qs, 72);
  v16h qa1 = frag_a_ld(Qs + 32, 72);

  int rbase = (lane >> 4) * 8, cn = lane & 15;
  float mi[8], mrun[8], lrun[8];
#pragma unroll
  for (int r = 0; r < 8; ++r) {
    mi[r] = mask[b * L_ + m0 + rbase + r];
    mrun[r] = -3.0e38f;
    lrun[r] = 0.f;
  }
  v8f zero = {};
  v8f oacc[4];
#pragma unroll
  for (int nt = 0; nt < 4; ++nt) oacc[nt] = zero;

  for (int j0 = 0; j0 < L_; j0 += 64) {
    __syncthreads();
#pragma unroll
    for (int i = 0; i < 4; ++i) {                      // stage 64x64 K and V
      int vi = t + i * 256;
      int jl = vi >> 4, c = (vi & 15) << 2;
      size_t g = (size_t)(b * L_ + j0 + jl) * D_ + h * DH_ + c;
      float4 kv = *(const float4*)&Kx[g];
      float4 vv = *(const float4*)&Vx[g];
      *(v4h*)&Ks[jl * 72 + c] = cvt4(kv);
      VsT[(c + 0) * 72 + jl] = (_Float16)vv.x;
      VsT[(c + 1) * 72 + jl] = (_Float16)vv.y;
      VsT[(c + 2) * 72 + jl] = (_Float16)vv.z;
      VsT[(c + 3) * 72 + jl] = (_Float16)vv.w;
    }
    __syncthreads();

    v8f st[4];
#pragma unroll
    for (int jt = 0; jt < 4; ++jt) {                   // S = Q K^T
      v8f s = zero;
      s = wmma32(qa0, frag_b_ld(&Ks[jt * 16 * 72], 72), s);
      s = wmma32(qa1, frag_b_ld(&Ks[jt * 16 * 72 + 32], 72), s);
      st[jt] = s;
    }
    float rowmax[8];
#pragma unroll
    for (int r = 0; r < 8; ++r) rowmax[r] = -3.0e38f;
#pragma unroll
    for (int jt = 0; jt < 4; ++jt) {                   // mask + row max
      float mj = mask[b * L_ + j0 + jt * 16 + cn];
#pragma unroll
      for (int r = 0; r < 8; ++r) {
        float mm = mi[r] * mj;
        float sv = st[jt][r] * mm + NEGV * (1.f - mm);
        st[jt][r] = sv;
        rowmax[r] = fmaxf(rowmax[r], sv);
      }
    }
#pragma unroll
    for (int d = 1; d < 16; d <<= 1)
#pragma unroll
      for (int r = 0; r < 8; ++r)
        rowmax[r] = fmaxf(rowmax[r], __shfl_xor(rowmax[r], d, 32));

    float corr[8], rsum[8];
#pragma unroll
    for (int r = 0; r < 8; ++r) {
      float mn = fmaxf(mrun[r], rowmax[r]);
      corr[r] = __expf(mrun[r] - mn);
      mrun[r] = mn;
      rsum[r] = 0.f;
    }
#pragma unroll
    for (int jt = 0; jt < 4; ++jt)
#pragma unroll
      for (int r = 0; r < 8; ++r) {
        float p = __expf(st[jt][r] - mrun[r]);
        st[jt][r] = p;
        rsum[r] += p;
      }
#pragma unroll
    for (int d = 1; d < 16; d <<= 1)
#pragma unroll
      for (int r = 0; r < 8; ++r) rsum[r] += __shfl_xor(rsum[r], d, 32);
#pragma unroll
    for (int r = 0; r < 8; ++r) lrun[r] = lrun[r] * corr[r] + rsum[r];
#pragma unroll
    for (int nt = 0; nt < 4; ++nt)
#pragma unroll
      for (int r = 0; r < 8; ++r) oacc[nt][r] *= corr[r];

    _Float16* Ps = Ws[wave];                           // P -> A fragments
#pragma unroll
    for (int jt = 0; jt < 4; ++jt)
#pragma unroll
      for (int r = 0; r < 8; ++r)
        Ps[(rbase + r) * 72 + jt * 16 + cn] = (_Float16)st[jt][r];
    asm volatile("s_wait_dscnt 0" ::: "memory");
    __builtin_amdgcn_wave_barrier();
    v16h pa0 = frag_a_ld(Ps, 72);
    v16h pa1 = frag_a_ld(Ps + 32, 72);
#pragma unroll
    for (int nt = 0; nt < 4; ++nt) {                   // O += P V
      oacc[nt] = wmma32(pa0, frag_b_ld(&VsT[nt * 16 * 72], 72), oacc[nt]);
      oacc[nt] = wmma32(pa1, frag_b_ld(&VsT[nt * 16 * 72 + 32], 72), oacc[nt]);
    }
  }
#pragma unroll
  for (int nt = 0; nt < 4; ++nt)
#pragma unroll
    for (int r = 0; r < 8; ++r) {
      int row = m0 + rbase + r;
      int col = h * DH_ + nt * 16 + cn;
      float dnm = lrun[r];
      O[(size_t)(b * L_ + row) * D_ + col] = (dnm > 0.f) ? oacc[nt][r] / dnm : 0.f;
    }
}

// ---------------------------------------------------------------- elementwise
__global__ __launch_bounds__(256) void k_add_pe(const float* __restrict__ X,
                                                const float* __restrict__ pe,
                                                float* __restrict__ Y) {
  size_t i = (size_t)blockIdx.x * 256 + threadIdx.x;
  Y[i] = X[i] + pe[i % LD_];
}

__global__ __launch_bounds__(256) void k_dwconv(const float* __restrict__ X,
                                                const float* __restrict__ w,
                                                const float* __restrict__ bias,
                                                float* __restrict__ Y) {
  size_t i = (size_t)blockIdx.x * 256 + threadIdx.x;
  int d = (int)(i % D_);
  int l = (int)((i / D_) % L_);
  size_t bbase = i - (size_t)l * D_ - d;
  float acc = bias[d];
#pragma unroll
  for (int tt = 0; tt < KW; ++tt) {
    int ls = l + tt - 3;
    if (ls >= 0 && ls < L_) acc += X[bbase + (size_t)ls * D_ + d] * w[d * KW + tt];
  }
  Y[i] = fmaxf(acc, 0.f);
}

__global__ __launch_bounds__(256) void k_stats(const float* __restrict__ X,
                                               float* __restrict__ mu,
                                               float* __restrict__ rs) {
  int b = blockIdx.x, t = threadIdx.x;
  const float* p = X + (size_t)b * LD_;
  float s = 0.f, s2 = 0.f;
  for (int i = t; i < LD_; i += 256) {
    float v = p[i];
    s += v;
    s2 += v * v;
  }
  __shared__ float sh[256], sh2[256];
  sh[t] = s; sh2[t] = s2;
  __syncthreads();
  for (int d = 128; d > 0; d >>= 1) {
    if (t < d) { sh[t] += sh[t + d]; sh2[t] += sh2[t + d]; }
    __syncthreads();
  }
  if (t == 0) {
    float m = sh[0] / (float)LD_;
    float var = sh2[0] / (float)LD_ - m * m;
    mu[b] = m;
    rs[b] = rsqrtf(var + 1e-5f);
  }
}

__global__ __launch_bounds__(256) void k_add_ln(
    const float* __restrict__ Z, const float* __restrict__ R,
    const float* __restrict__ g, const float* __restrict__ be,
    const float* __restrict__ mu, const float* __restrict__ rs,
    float* __restrict__ Y) {
  size_t i = (size_t)blockIdx.x * 256 + threadIdx.x;
  int b = (int)(i / LD_);
  int ld = (int)(i % LD_);
  Y[i] = Z[i] + (R[i] - mu[b]) * rs[b] * g[ld] + be[ld];
}

// ---------------------------------------------------------------- weight pack
// pw_w/fc_w/out_w are already [N][K] row-major: straight f32 -> f16 convert.
__global__ __launch_bounds__(256) void k_pack_cvt(const float* __restrict__ W,
                                                  _Float16* __restrict__ Bp) {
  int i = blockIdx.x * 256 + threadIdx.x;      // over N*K/4, vectorized
  float4 v = *(const float4*)&W[(size_t)i * 4];
  *(v4h*)&Bp[(size_t)i * 4] = cvt4(v);
}

// wq[h,d,dh] -> Bp[j][d] with j = h*64+dh (N-major, K = d).
__global__ __launch_bounds__(256) void k_pack_qkv(const float* __restrict__ W,
                                                  _Float16* __restrict__ Bp) {
  int i = blockIdx.x * 256 + threadIdx.x;      // i = j*512 + d
  int j = i >> 9, d = i & 511;
  int h = j >> 6, dh = j & 63;
  Bp[i] = (_Float16)W[((size_t)h * D_ + d) * DH_ + dh];
}

// ---------------------------------------------------------------- launcher
extern "C" void kernel_launch(void* const* d_in, const int* in_sizes, int n_in,
                              void* d_out, int out_size, void* d_ws,
                              size_t ws_size, hipStream_t stream) {
  const float* x     = (const float*)d_in[0];
  const float* xmask = (const float*)d_in[1];
  const float* pe    = (const float*)d_in[2];
  const float* dw_w  = (const float*)d_in[3];
  const float* dw_b  = (const float*)d_in[4];
  const float* pw_w  = (const float*)d_in[5];
  const float* pw_b  = (const float*)d_in[6];
  const float* wq    = (const float*)d_in[7];
  const float* bq    = (const float*)d_in[8];
  const float* wk    = (const float*)d_in[9];
  const float* bk    = (const float*)d_in[10];
  const float* wv    = (const float*)d_in[11];
  const float* bv    = (const float*)d_in[12];
  const float* fc_w  = (const float*)d_in[13];
  const float* fc_b  = (const float*)d_in[14];
  const float* out_w = (const float*)d_in[15];
  const float* out_b = (const float*)d_in[16];
  const float* lnc_g = (const float*)d_in[17];
  const float* lnc_b = (const float*)d_in[18];
  const float* lna_g = (const float*)d_in[19];
  const float* lna_b = (const float*)d_in[20];
  const float* lno_g = (const float*)d_in[21];
  const float* lno_b = (const float*)d_in[22];
  float* out = (float*)d_out;

  char* wp = (char*)d_ws;
  auto alloc = [&](size_t bytes) -> void* {
    void* p = (void*)wp;
    wp += (bytes + 255) & ~(size_t)255;
    return p;
  };
  float* x0 = (float*)alloc(BLD * 4);
  float* t1 = (float*)alloc(BLD * 4);
  float* t2 = (float*)alloc(BLD * 4);
  float* qb = (float*)alloc(BLD * 4);
  float* kb = (float*)alloc(BLD * 4);
  float* vb = (float*)alloc(BLD * 4);
  float* ob = (float*)alloc(BLD * 4);
  _Float16* pwp  = (_Float16*)alloc((size_t)3 * D_ * D_ * 2);
  _Float16* qp   = (_Float16*)alloc((size_t)D_ * D_ * 2);
  _Float16* kp   = (_Float16*)alloc((size_t)D_ * D_ * 2);
  _Float16* vp   = (_Float16*)alloc((size_t)D_ * D_ * 2);
  _Float16* fcp  = (_Float16*)alloc((size_t)D_ * D_ * 2);
  _Float16* outp = (_Float16*)alloc((size_t)D_ * D_ * 2);
  float* mu = (float*)alloc(B_ * 4);
  float* rs = (float*)alloc(B_ * 4);

  int egrid = (int)(BLD / 256);
  int pgrid = (D_ * D_) / 256;          // scalar pack grid
  int pgrid4 = (D_ * D_ / 4) / 256;     // vectorized pack grid
  dim3 ggrid(B_ * L_ / 128, D_ / 128);  // (128, 4)
  int M = B_ * L_;

  for (int i = 0; i < 3; ++i)
    k_pack_cvt<<<pgrid4, 256, 0, stream>>>(pw_w + (size_t)i * D_ * D_,
                                           pwp + (size_t)i * D_ * D_);
  k_pack_qkv<<<pgrid, 256, 0, stream>>>(wq, qp);
  k_pack_qkv<<<pgrid, 256, 0, stream>>>(wk, kp);
  k_pack_qkv<<<pgrid, 256, 0, stream>>>(wv, vp);
  k_pack_cvt<<<pgrid4, 256, 0, stream>>>(fc_w, fcp);
  k_pack_cvt<<<pgrid4, 256, 0, stream>>>(out_w, outp);

  k_add_pe<<<egrid, 256, 0, stream>>>(x, pe, x0);

  // conv stack with rotating buffers
  float* xc = x0; float* ta = t1; float* tb = t2;
  for (int i = 0; i < 3; ++i) {
    k_dwconv<<<egrid, 256, 0, stream>>>(xc, dw_w + (size_t)i * D_ * KW,
                                        dw_b + (size_t)i * D_, ta);
    k_gemm<<<ggrid, 256, 0, stream>>>(ta, pwp + (size_t)i * D_ * D_,
                                      pw_b + (size_t)i * D_, tb, M, D_, D_, 1);
    if (i == 0) {
      float* tmp = xc; xc = tb; tb = tmp;
    } else {
      k_stats<<<B_, 256, 0, stream>>>(xc, mu, rs);
      k_add_ln<<<egrid, 256, 0, stream>>>(tb, xc, lnc_g + (size_t)(i - 1) * LD_,
                                          lnc_b + (size_t)(i - 1) * LD_, mu, rs,
                                          ta);
      float* tmp = xc; xc = ta; ta = tmp;
    }
  }

  // attention: xc is the residual input
  k_gemm<<<ggrid, 256, 0, stream>>>(xc, qp, bq, qb, M, D_, D_, 0);
  k_gemm<<<ggrid, 256, 0, stream>>>(xc, kp, bk, kb, M, D_, D_, 0);
  k_gemm<<<ggrid, 256, 0, stream>>>(xc, vp, bv, vb, M, D_, D_, 0);
  k_flash<<<dim3(B_ * H_, L_ / 128), 256, 0, stream>>>(qb, kb, vb, xmask, ob);
  k_gemm<<<ggrid, 256, 0, stream>>>(ob, fcp, fc_b, ta, M, D_, D_, 0);
  k_stats<<<B_, 256, 0, stream>>>(xc, mu, rs);
  k_add_ln<<<egrid, 256, 0, stream>>>(ta, xc, lna_g, lna_b, mu, rs, tb);

  // output linear: tb is the residual input
  k_gemm<<<ggrid, 256, 0, stream>>>(tb, outp, out_b, ob, M, D_, D_, 0);
  k_stats<<<B_, 256, 0, stream>>>(tb, mu, rs);
  k_add_ln<<<egrid, 256, 0, stream>>>(ob, tb, lno_g, lno_b, mu, rs, out);
}
